// UserAgg_21354577396099
// MI455X (gfx1250) — matrix operand order repaired and checked by reading, not compiled, not run on hardware
//
#include <hip/hip_runtime.h>

typedef __attribute__((ext_vector_type(16))) __bf16         v16bf;
typedef __attribute__((ext_vector_type(8)))  float          v8f;
typedef __attribute__((ext_vector_type(8)))  unsigned short ushort8;

#define DEV static __device__ __forceinline__
#define NW 4   // waves per block

// ---------- scalar helpers ----------
DEV unsigned short f2bf(float f) {               // fp32 -> bf16 (RNE)
  unsigned u = __float_as_uint(f);
  u += 0x7FFFu + ((u >> 16) & 1u);
  return (unsigned short)(u >> 16);
}
DEV float bf2f(unsigned short h) { return __uint_as_float(((unsigned)h) << 16); }
// order-preserving float<->uint encoding for atomicMax-based segment max
DEV unsigned ford(float f)   { unsigned u = __float_as_uint(f); return (u & 0x80000000u) ? ~u : (u | 0x80000000u); }
DEV float    funord(unsigned e) { unsigned u = (e & 0x80000000u) ? (e & 0x7FFFFFFFu) : ~e; return __uint_as_float(u); }

DEV v16bf ld_frag(const unsigned short* p0, const unsigned short* p1) {
  union { v16bf v; ushort8 h[2]; } u;
  u.h[0] = *(const ushort8*)p0;
  u.h[1] = *(const ushort8*)p1;
  return u.v;
}

// ---------------------------------------------------------------------------
// out[16x64] = act(in[16xK] @ W[64xK]^T + bias), all bf16 with f32 accum.
// A-fragments are hoisted into registers once and reused across all 4 N-tiles
// (4x less LDS read traffic than reloading per tile).
// A layout: lane->M=lane&15; elem j -> K = 32c + j + 8*(j>=8) + 8*(lane>=16)
// B layout: lane->N=lane&15; elem j -> K = 32c + j + 8*(j>=8) + 8*(lane>=16)
// D layout: lane->N=lane&15; vgpr r -> M = r + 8*(lane>=16)
// ---------------------------------------------------------------------------
template <int K>
DEV void tile_gemm(const unsigned short* in,
                   const unsigned short* __restrict__ W,
                   const float* __restrict__ bias,
                   unsigned short* out, int outStride,
                   unsigned short* __restrict__ gout, int gRows,
                   int lane, bool relu)
{
  const int n0 = lane & 15;
  const int hf = lane >> 4;
  constexpr int KC = K >> 5;
  v16bf Afr[KC];
#pragma unroll
  for (int c = 0; c < KC; ++c) {
    const unsigned short* ap = in + n0 * K + 32 * c + 8 * hf;
    Afr[c] = ld_frag(ap, ap + 16);
  }
#pragma unroll
  for (int t = 0; t < 4; ++t) {
    const float bv = bias[16 * t + n0];
    v8f acc;
#pragma unroll
    for (int r = 0; r < 8; ++r) acc[r] = bv;
#pragma unroll
    for (int c = 0; c < KC; ++c) {
      const unsigned short* wp = W + (size_t)(16 * t + n0) * K + 32 * c + 8 * hf;
      v16bf B = ld_frag(wp, wp + 16);
      acc = __builtin_amdgcn_wmma_f32_16x16x32_bf16(
                false, Afr[c], false, B, (short)0, acc, false, false);
    }
#pragma unroll
    for (int r = 0; r < 8; ++r) {
      float v = acc[r];
      if (relu) v = v > 0.0f ? v : 0.0f;
      const int mr = r + 8 * hf;
      const unsigned short hv = f2bf(v);
      out[mr * outStride + 16 * t + n0] = hv;
      if (gout && mr < gRows) gout[(size_t)mr * 64 + 16 * t + n0] = hv;
    }
  }
}

// ---------------------------------------------------------------------------
// Kernel 0: fp32 -> bf16 weight conversion into workspace
// ---------------------------------------------------------------------------
__global__ void cvt_kernel(const float* __restrict__ src,
                           unsigned short* __restrict__ dst, int n) {
  const int i = blockIdx.x * blockDim.x + threadIdx.x;
  if (i < n) dst[i] = f2bf(src[i]);
}

// ---------------------------------------------------------------------------
// Kernel 1: init z / denom / segment-max buffers
// ---------------------------------------------------------------------------
__global__ void init_kernel(float* __restrict__ z, float* __restrict__ denom,
                            unsigned* __restrict__ imax, int item_num) {
  const size_t i = (size_t)blockIdx.x * blockDim.x + threadIdx.x;
  if (i < (size_t)item_num * 64) z[i] = 0.0f;
  if (i < (size_t)item_num) { denom[i] = 0.0f; imax[i] = 0x007FFFFFu; /* ford(-inf) */ }
}

// ---------------------------------------------------------------------------
// Kernel 2: per-edge MLP chain + attention score, one wave per 16-edge tile.
// ---------------------------------------------------------------------------
__global__ __launch_bounds__(128) void edge_mlp_kernel(
    const float* __restrict__ user_feat, const float* __restrict__ item_feat,
    const float* __restrict__ rating_table,
    const float* __restrict__ gu_b1, const float* __restrict__ gu_b2,
    const float* __restrict__ att_b1, const float* __restrict__ att_b2,
    const float* __restrict__ att_w3, const float* __restrict__ att_b3,
    const int* __restrict__ row_idx, const int* __restrict__ col_idx,
    const int* __restrict__ rating, const int* __restrict__ user_num_p,
    const unsigned short* __restrict__ wb1, const unsigned short* __restrict__ wb2,
    const unsigned short* __restrict__ wa1, const unsigned short* __restrict__ wa2,
    unsigned short* __restrict__ fjt, float* __restrict__ score,
    unsigned* __restrict__ imax, int E)
{
  __shared__ __align__(16) unsigned short sBuf[NW][16 * 128];
  __shared__ __align__(16) unsigned short sTmp[NW][16 * 64];
  const int lane = threadIdx.x & 31;
  const int wid  = threadIdx.x >> 5;
  const int m    = lane & 15;
  const int hf   = lane >> 4;
  const int user_num = user_num_p[0];
  unsigned short* xb = sBuf[wid];
  unsigned short* tb = sTmp[wid];
  const int ntiles = (E + 15) >> 4;
  const int wstep  = gridDim.x * NW;

  for (int tile = blockIdx.x * NW + wid; tile < ntiles; tile += wstep) {
    const int ebase = tile * 16;
    // ---- stage x = [u_emb | r_emb] (bf16) into LDS: lane<16 -> u, lane>=16 -> r
    {
      int e = ebase + m; if (e >= E) e = E - 1;
      const float* src = hf ? (rating_table + (size_t)rating[e] * 64)
                            : (user_feat   + (size_t)row_idx[e] * 64);
      unsigned short* dst = xb + m * 128 + hf * 64;
#pragma unroll
      for (int k = 0; k < 64; k += 4) {
        const float4 v = *(const float4*)(src + k);
        dst[k + 0] = f2bf(v.x); dst[k + 1] = f2bf(v.y);
        dst[k + 2] = f2bf(v.z); dst[k + 3] = f2bf(v.w);
      }
    }
    asm volatile("s_wait_dscnt 0" ::: "memory");
    // h = relu(x @ gu_w1^T + b1)  [16,128]->[16,64]
    tile_gemm<128>(xb, wb1, gu_b1, tb, 64, nullptr, 0, lane, true);
    asm volatile("s_wait_dscnt 0" ::: "memory");
    // f = relu(h @ gu_w2^T + b2)  -> xb[:,0:64] and global fjt
    tile_gemm<64>(tb, wb2, gu_b2, xb, 128,
                  fjt + (size_t)ebase * 64, E - ebase, lane, true);
    // ---- stage i_emb (bf16) into xb[:,64:128]; each lane does half a row
    {
      int e = ebase + m; if (e >= E) e = E - 1;
      const float* src = item_feat + (size_t)(col_idx[e] + user_num) * 64 + hf * 32;
      unsigned short* dst = xb + m * 128 + 64 + hf * 32;
#pragma unroll
      for (int k = 0; k < 32; k += 4) {
        const float4 v = *(const float4*)(src + k);
        dst[k + 0] = f2bf(v.x); dst[k + 1] = f2bf(v.y);
        dst[k + 2] = f2bf(v.z); dst[k + 3] = f2bf(v.w);
      }
    }
    asm volatile("s_wait_dscnt 0" ::: "memory");
    // a1 = relu([f ; i] @ att_w1^T + ab1) -> tb
    tile_gemm<128>(xb, wa1, att_b1, tb, 64, nullptr, 0, lane, true);
    asm volatile("s_wait_dscnt 0" ::: "memory");
    // a2 = relu(a1 @ att_w2^T + ab2) -> xb[:,64:128]
    tile_gemm<64>(tb, wa2, att_b2, xb + 64, 128, nullptr, 0, lane, true);
    asm volatile("s_wait_dscnt 0" ::: "memory");
    // score[e] = a2 . att_w3 + b3 ; lanes 0..15 each own one edge
    if (lane < 16) {
      const int e = ebase + lane;
      if (e < E) {
        float s = att_b3[0];
        const unsigned short* ar = xb + lane * 128 + 64;
#pragma unroll 16
        for (int k = 0; k < 64; ++k) s += bf2f(ar[k]) * att_w3[k];
        score[e] = s;
        atomicMax(&imax[col_idx[e]], ford(s));
      }
    }
    asm volatile("s_wait_dscnt 0" ::: "memory");
  }
}

// ---------------------------------------------------------------------------
// Kernel 3: e = exp(score - segmax), denominator accumulation
// ---------------------------------------------------------------------------
__global__ void softmax_kernel(const float* __restrict__ score,
                               const int* __restrict__ col_idx,
                               const unsigned* __restrict__ imax,
                               float* __restrict__ expv,
                               float* __restrict__ denom, int E) {
  const int e = blockIdx.x * blockDim.x + threadIdx.x;
  if (e >= E) return;
  const int c = col_idx[e];
  const float ev = __expf(score[e] - funord(imax[c]));
  expv[e] = ev;
  atomicAdd(&denom[c], ev);
}

// ---------------------------------------------------------------------------
// Kernel 4: z[col] += f_jt * (e/denom)  (4 threads per edge, 16 floats each)
// ---------------------------------------------------------------------------
__global__ void scatter_kernel(const unsigned short* __restrict__ fjt,
                               const float* __restrict__ expv,
                               const float* __restrict__ denom,
                               const int* __restrict__ col_idx,
                               float* __restrict__ z, int E) {
  const int g = blockIdx.x * blockDim.x + threadIdx.x;
  const int e = g >> 2;
  if (e >= E) return;
  const int seg = (g & 3) * 16;
  const int c = col_idx[e];
  const float v = expv[e] / denom[c];
  const unsigned short* fr = fjt + (size_t)e * 64 + seg;
  float* zr = z + (size_t)c * 64 + seg;
#pragma unroll
  for (int i = 0; i < 16; ++i) atomicAdd(&zr[i], bf2f(fr[i]) * v);
}

// ---------------------------------------------------------------------------
// Kernel 5: out = z @ w_w^T + w_b   (WMMA, one wave per 16-row tile)
// ---------------------------------------------------------------------------
__global__ __launch_bounds__(128) void out_gemm_kernel(
    const float* __restrict__ z, const unsigned short* __restrict__ wbw,
    const float* __restrict__ w_b, float* __restrict__ out, int item_num)
{
  __shared__ __align__(16) unsigned short sZ[NW][16 * 64];
  const int lane = threadIdx.x & 31;
  const int wid  = threadIdx.x >> 5;
  const int m    = lane & 15;
  const int hf   = lane >> 4;
  unsigned short* zb = sZ[wid];
  const int ntiles = (item_num + 15) >> 4;
  const int wstep  = gridDim.x * NW;
  for (int tile = blockIdx.x * NW + wid; tile < ntiles; tile += wstep) {
    const int row = tile * 16 + m;
    {
      const bool ok = row < item_num;
      const float* src = z + (size_t)(ok ? row : 0) * 64 + hf * 32;
      unsigned short* dst = zb + m * 64 + hf * 32;
#pragma unroll
      for (int k = 0; k < 32; k += 4) {
        float4 v = *(const float4*)(src + k);
        if (!ok) { v.x = 0.f; v.y = 0.f; v.z = 0.f; v.w = 0.f; }
        dst[k + 0] = f2bf(v.x); dst[k + 1] = f2bf(v.y);
        dst[k + 2] = f2bf(v.z); dst[k + 3] = f2bf(v.w);
      }
    }
    asm volatile("s_wait_dscnt 0" ::: "memory");
    v16bf Afr[2];
#pragma unroll
    for (int c = 0; c < 2; ++c) {
      const unsigned short* ap = zb + m * 64 + 32 * c + 8 * hf;
      Afr[c] = ld_frag(ap, ap + 16);
    }
#pragma unroll
    for (int t = 0; t < 4; ++t) {
      const float bv = w_b[16 * t + m];
      v8f acc;
#pragma unroll
      for (int r = 0; r < 8; ++r) acc[r] = bv;
#pragma unroll
      for (int c = 0; c < 2; ++c) {
        const unsigned short* wp = wbw + (size_t)(16 * t + m) * 64 + 32 * c + 8 * hf;
        v16bf B = ld_frag(wp, wp + 16);
        acc = __builtin_amdgcn_wmma_f32_16x16x32_bf16(
                  false, Afr[c], false, B, (short)0, acc, false, false);
      }
#pragma unroll
      for (int r = 0; r < 8; ++r) {
        const int mr = tile * 16 + r + 8 * hf;
        if (mr < item_num) out[(size_t)mr * 64 + 16 * t + m] = acc[r];
      }
    }
    asm volatile("s_wait_dscnt 0" ::: "memory");
  }
}

// ---------------------------------------------------------------------------
extern "C" void kernel_launch(void* const* d_in, const int* in_sizes, int n_in,
                              void* d_out, int out_size, void* d_ws, size_t ws_size,
                              hipStream_t stream) {
  const float* user_feat  = (const float*)d_in[0];
  const float* item_feat  = (const float*)d_in[1];
  const float* rating_tab = (const float*)d_in[2];
  const float* gu_w1  = (const float*)d_in[3];
  const float* gu_b1  = (const float*)d_in[4];
  const float* gu_w2  = (const float*)d_in[5];
  const float* gu_b2  = (const float*)d_in[6];
  const float* att_w1 = (const float*)d_in[7];
  const float* att_b1 = (const float*)d_in[8];
  const float* att_w2 = (const float*)d_in[9];
  const float* att_b2 = (const float*)d_in[10];
  const float* att_w3 = (const float*)d_in[11];
  const float* att_b3 = (const float*)d_in[12];
  const float* w_w    = (const float*)d_in[13];
  const float* w_b    = (const float*)d_in[14];
  const int* row_idx  = (const int*)d_in[15];
  const int* col_idx  = (const int*)d_in[16];
  const int* rating   = (const int*)d_in[17];
  const int* user_num_p = (const int*)d_in[18];
  const int E = in_sizes[15];
  const int item_num = out_size / 64;
  float* outf = (float*)d_out;
  (void)n_in; (void)ws_size;

  char* ws = (char*)d_ws;
  size_t off = 0;
  auto take = [&](size_t bytes) -> void* {
    void* p = (void*)(ws + off);
    off = (off + bytes + 255) & ~(size_t)255;
    return p;
  };
  unsigned short* wb1   = (unsigned short*)take(64 * 128 * 2);
  unsigned short* wb2   = (unsigned short*)take(64 * 64 * 2);
  unsigned short* wa1   = (unsigned short*)take(64 * 128 * 2);
  unsigned short* wa2   = (unsigned short*)take(64 * 64 * 2);
  unsigned short* wbw   = (unsigned short*)take(64 * 64 * 2);
  unsigned*       imax  = (unsigned*)take((size_t)item_num * 4);
  float*          denom = (float*)take((size_t)item_num * 4);
  float*          zbuf  = (float*)take((size_t)item_num * 64 * 4);
  float*          score = (float*)take((size_t)E * 4);
  float*          expv  = (float*)take((size_t)E * 4);
  unsigned short* fjt   = (unsigned short*)take((size_t)E * 64 * 2);

  cvt_kernel<<<(64 * 128 + 255) / 256, 256, 0, stream>>>(gu_w1, wb1, 64 * 128);
  cvt_kernel<<<(64 * 64 + 255) / 256, 256, 0, stream>>>(gu_w2, wb2, 64 * 64);
  cvt_kernel<<<(64 * 128 + 255) / 256, 256, 0, stream>>>(att_w1, wa1, 64 * 128);
  cvt_kernel<<<(64 * 64 + 255) / 256, 256, 0, stream>>>(att_w2, wa2, 64 * 64);
  cvt_kernel<<<(64 * 64 + 255) / 256, 256, 0, stream>>>(w_w, wbw, 64 * 64);
  init_kernel<<<(int)(((size_t)item_num * 64 + 255) / 256), 256, 0, stream>>>(
      zbuf, denom, imax, item_num);

  const int ntiles = (E + 15) / 16;
  int grid1 = (ntiles + NW - 1) / NW;
  if (grid1 > 4096) grid1 = 4096;
  edge_mlp_kernel<<<grid1, 128, 0, stream>>>(
      user_feat, item_feat, rating_tab, gu_b1, gu_b2, att_b1, att_b2,
      att_w3, att_b3, row_idx, col_idx, rating, user_num_p,
      wb1, wb2, wa1, wa2, fjt, score, imax, E);

  softmax_kernel<<<(E + 255) / 256, 256, 0, stream>>>(score, col_idx, imax,
                                                      expv, denom, E);
  scatter_kernel<<<(int)(((size_t)E * 4 + 255) / 256), 256, 0, stream>>>(
      fjt, expv, denom, col_idx, zbuf, E);

  const int ntiles2 = (item_num + 15) / 16;
  int grid2 = (ntiles2 + NW - 1) / NW;
  if (grid2 > 2048) grid2 = 2048;
  out_gemm_kernel<<<grid2, 128, 0, stream>>>(zbuf, wbw, w_b, outf, item_num);
}
